// IDMForwardSim_16466904612948
// MI455X (gfx1250) — matrix operand order repaired and checked by reading, not compiled, not run on hardware
//
#include <hip/hip_runtime.h>
#include <math.h>

typedef __attribute__((ext_vector_type(16))) _Float16 v16h;
typedef __attribute__((ext_vector_type(8)))  float    v8f;

union HFrag { v16h v; _Float16 h[16]; unsigned u[8]; };
union FAcc  { v8f  v; float f[8]; };

#define TILE_M 16
#define HID    128
#define NENV   9

__global__ __launch_bounds__(256, 1) void rollout_kernel(
    const float* __restrict__ idm_params,
    const float* __restrict__ proj_belief,
    const float* __restrict__ enc_h,
    const float* __restrict__ idm_s,
    const float* __restrict__ merger_cs,
    const float* __restrict__ W1, const float* __restrict__ b1,
    const float* __restrict__ W2, const float* __restrict__ b2,
    const float* __restrict__ W3, const float* __restrict__ b3,
    const float* __restrict__ Wf, const float* __restrict__ bf_,
    const float* __restrict__ Wm, const float* __restrict__ bm_,
    const float* __restrict__ smean, const float* __restrict__ svar,
    float* __restrict__ out, int B, int T)
{
    __shared__ float     sBase[TILE_M * HID];   // time-invariant L1 preact (b1 + belief/enc part - mean/std term)
    __shared__ float     sW1s[NENV * HID];      // env rows of W1 pre-scaled by 1/std (mc rows raw)
    __shared__ _Float16  sCoefH[TILE_M * 32];   // per-step 16x32 A-matrix for layer 1 (K padded w/ zeros)
    __shared__ _Float16  sH1[TILE_M * HID];
    __shared__ _Float16  sH2[TILE_M * HID];
    __shared__ _Float16  sH3[TILE_M * HID];
    __shared__ float     sLog[32];              // [0..15]=f logits, [16..31]=m logits

    const int tid  = threadIdx.x;
    const int lane = tid & 31;
    const int wave = tid >> 5;
    const int half = (lane >> 4) & 1;
    const int ln   = lane & 15;
    const int b0   = blockIdx.x * TILE_M;
    const int nBase = wave * 16;
    const int col   = nBase + ln;

    // ---- one-time: zero padded coef matrix; env/mc rows of W1 (scaled) ----
    ((unsigned*)sCoefH)[tid] = 0u;              // 256 dwords == 16x32 halves
    for (int idx = tid; idx < NENV * HID; idx += 256) {
        int i = idx / HID, j = idx - i * HID;
        float w = W1[(192 + i) * HID + j];
        if (i < 6) w *= rsqrtf(svar[i]);
        sW1s[idx] = w;
    }
    __syncthreads();

    // ---- one-time: base1c = b1 - (mean/std)·W1env + [belief,enc] @ W1[0:192] ----
    {
        const int r = tid >> 4;      // 0..15 batch row in tile
        const int c = tid & 15;      // 0..15 column phase
        float acc[8];
        #pragma unroll
        for (int e = 0; e < 8; ++e) {
            int j = c + 16 * e;
            float a = b1[j];
            for (int i = 0; i < 6; ++i) a -= smean[i] * sW1s[i * HID + j];
            acc[e] = a;
        }
        const float* pb = proj_belief + (size_t)(b0 + r) * 64;
        const float* eh = enc_h       + (size_t)(b0 + r) * HID;
        for (int k = 0; k < 192; ++k) {
            float x = (k < 64) ? pb[k] : eh[k - 64];
            const float* wrow = W1 + (size_t)k * HID + c;
            #pragma unroll
            for (int e = 0; e < 8; ++e) acc[e] += x * wrow[16 * e];
        }
        #pragma unroll
        for (int e = 0; e < 8; ++e) sBase[r * HID + c + 16 * e] = acc[e];
    }

    // ---- one-time: per-wave resident f16 B-fragments ----
    HFrag w1f, w2f[4], w3f[4], whf[4];
    #pragma unroll
    for (int v = 0; v < 8; ++v) {               // layer-1 env weights (K rows >= 9 are zero)
        int kb = (v >> 2) * 16 + half * 8 + (v & 3) * 2;
        w1f.h[2*v  ] = (_Float16)((kb     < NENV) ? sW1s[kb * HID + col]       : 0.0f);
        w1f.h[2*v+1] = (_Float16)((kb + 1 < NENV) ? sW1s[(kb + 1) * HID + col] : 0.0f);
    }
    for (int kt = 0; kt < 4; ++kt) {
        #pragma unroll
        for (int v = 0; v < 8; ++v) {
            int kb = kt * 32 + (v >> 2) * 16 + half * 8 + (v & 3) * 2;
            w2f[kt].h[2*v  ] = (_Float16)W2[(size_t)kb * HID + col];
            w2f[kt].h[2*v+1] = (_Float16)W2[(size_t)(kb + 1) * HID + col];
            w3f[kt].h[2*v  ] = (_Float16)W3[(size_t)kb * HID + col];
            w3f[kt].h[2*v+1] = (_Float16)W3[(size_t)(kb + 1) * HID + col];
            // head matrix: col 0 = Wf, col 1 = Wm, cols 2..15 = 0
            float h0 = (ln == 0) ? Wf[kb]     : (ln == 1) ? Wm[kb]     : 0.0f;
            float h1v = (ln == 0) ? Wf[kb + 1] : (ln == 1) ? Wm[kb + 1] : 0.0f;
            whf[kt].h[2*v  ] = (_Float16)h0;
            whf[kt].h[2*v+1] = (_Float16)h1v;
        }
    }
    const float bias2 = b2[col];
    const float bias3 = b3[col];

    // ---- rollout state: wave 0, lanes 0..15 each own one batch row ----
    float ego_v = 0.f, ego_x = 0.f;
    float des_v = 1.f, des_tgap = 0.f, min_jamx = 0.f, max_act = 1.f, gap_denum = 1.f;
    float bfs = 0.f, bms = 0.f;
    if (wave == 0 && lane < 16) {
        const float* ip = idm_params + (size_t)(b0 + lane) * 5;
        des_v = ip[0]; des_tgap = ip[1]; min_jamx = ip[2]; max_act = ip[3];
        gap_denum = 2.0f * sqrtf(max_act * ip[4]);
        const float* s0 = idm_s + (size_t)(b0 + lane) * T * 11;
        ego_v = s0[0]; ego_x = s0[3];
        bfs = bf_[0]; bms = bm_[0];
    }
    __syncthreads();

    // ---- one-time: time-invariant layer-1 C fragment lives in registers ----
    float baseC[8];
    #pragma unroll
    for (int v = 0; v < 8; ++v) baseC[v] = sBase[(v + 8 * half) * HID + col];

    const size_t BT = (size_t)B * T;
    float efdv = 0.f, efdx = 0.f, emdv = 0.f, emdx = 0.f;

    for (int t = 0; t < T; ++t) {
        // (a) env coefficients (raw; scaling folded into sW1s / baseC)
        if (wave == 0 && lane < 16) {
            const float* sp = idm_s + ((size_t)(b0 + lane) * T + t) * 11;
            float fvv = sp[1], mv = sp[2], fx = sp[4], mx = sp[5], mex = sp[10];
            efdv = ego_v - fvv;
            efdx = fx - ego_x;
            emdv = (ego_v - mv) * mex;                        // dummy em_dv = 0
            emdx = (mx - ego_x) * mex + (1.0f - mex) * 70.0f; // dummy em_dx = 70
            const float* mc = merger_cs + ((size_t)(b0 + lane) * T + t) * 3;
            _Float16* cf = sCoefH + lane * 32;
            cf[0] = (_Float16)ego_v; cf[1] = (_Float16)fvv;   cf[2] = (_Float16)efdv;
            cf[3] = (_Float16)efdx;  cf[4] = (_Float16)emdv;  cf[5] = (_Float16)emdx;
            cf[6] = (_Float16)mc[0]; cf[7] = (_Float16)mc[1]; cf[8] = (_Float16)mc[2];
        }
        __syncthreads();

        // (b) layer 1: single WMMA (K=32, coef x W1env), C = preloaded base fragment
        {
            HFrag a;
            #pragma unroll
            for (int v = 0; v < 8; ++v) {
                int kb = (v >> 2) * 16 + half * 8 + (v & 3) * 2;
                a.u[v] = *(const unsigned*)&sCoefH[ln * 32 + kb];
            }
            FAcc acc;
            #pragma unroll
            for (int v = 0; v < 8; ++v) acc.f[v] = baseC[v];
            acc.v = __builtin_amdgcn_wmma_f32_16x16x32_f16(
                false, a.v, false, w1f.v, (short)0, acc.v, false, false);
            #pragma unroll
            for (int v = 0; v < 8; ++v) {
                float s = acc.f[v];
                s = fmaxf(s, 0.3f * s);                       // leaky_relu(0.3)
                sH1[(v + 8 * half) * HID + col] = (_Float16)s;
            }
        }
        __syncthreads();

        // (c) layer 2: 16x16 tile per wave, K=128 via 4x wmma
        {
            FAcc acc;
            #pragma unroll
            for (int v = 0; v < 8; ++v) acc.f[v] = bias2;
            for (int kt = 0; kt < 4; ++kt) {
                HFrag a;
                #pragma unroll
                for (int v = 0; v < 8; ++v) {
                    int kb = kt * 32 + (v >> 2) * 16 + half * 8 + (v & 3) * 2;
                    a.u[v] = *(const unsigned*)&sH1[ln * HID + kb];
                }
                acc.v = __builtin_amdgcn_wmma_f32_16x16x32_f16(
                    false, a.v, false, w2f[kt].v, (short)0, acc.v, false, false);
            }
            #pragma unroll
            for (int v = 0; v < 8; ++v) {
                float s = acc.f[v];
                s = fmaxf(s, 0.3f * s);
                sH2[(v + 8 * half) * HID + col] = (_Float16)s;
            }
        }
        __syncthreads();

        // (d) layer 3: same shape
        {
            FAcc acc;
            #pragma unroll
            for (int v = 0; v < 8; ++v) acc.f[v] = bias3;
            for (int kt = 0; kt < 4; ++kt) {
                HFrag a;
                #pragma unroll
                for (int v = 0; v < 8; ++v) {
                    int kb = kt * 32 + (v >> 2) * 16 + half * 8 + (v & 3) * 2;
                    a.u[v] = *(const unsigned*)&sH2[ln * HID + kb];
                }
                acc.v = __builtin_amdgcn_wmma_f32_16x16x32_f16(
                    false, a.v, false, w3f[kt].v, (short)0, acc.v, false, false);
            }
            #pragma unroll
            for (int v = 0; v < 8; ++v) {
                float s = acc.f[v];
                s = fmaxf(s, 0.3f * s);
                sH3[(v + 8 * half) * HID + col] = (_Float16)s;
            }
        }
        __syncthreads();

        // (e) heads via WMMA on wave 0 only: [16x128] x [128x2] (cols 0/1 = Wf/Wm)
        if (wave == 0) {
            FAcc acc;
            #pragma unroll
            for (int v = 0; v < 8; ++v) acc.f[v] = 0.0f;
            for (int kt = 0; kt < 4; ++kt) {
                HFrag a;
                #pragma unroll
                for (int v = 0; v < 8; ++v) {
                    int kb = kt * 32 + (v >> 2) * 16 + half * 8 + (v & 3) * 2;
                    a.u[v] = *(const unsigned*)&sH3[ln * HID + kb];
                }
                acc.v = __builtin_amdgcn_wmma_f32_16x16x32_f16(
                    false, a.v, false, whf[kt].v, (short)0, acc.v, false, false);
            }
            if (ln < 2) {
                #pragma unroll
                for (int v = 0; v < 8; ++v)
                    sLog[ln * 16 + v + 8 * half] = acc.f[v];
            }
            // same-wave LDS ordering: compiler inserts s_wait_dscnt before reads below

            // (f) attention + IDM physics + state update + outputs
            if (lane < 16) {
                float df = bfs + sLog[lane];
                float dm = bms + sLog[16 + lane];
                float fs = __expf(5.0f * df);
                float ms = __expf(5.0f * dm);
                float inv = 1.0f / (fs + ms);
                float fatt = fs * inv, matt = ms * inv;

                float q = ego_v / des_v; float q2 = q * q; float q4 = q2 * q2;
                // front vehicle IDM
                float dxp = 0.1f + fmaxf(efdx, 0.0f);
                float dg  = fmaxf(min_jamx + des_tgap * ego_v + ego_v * efdv / gap_denum, 0.0f);
                float g   = dg / dxp;
                float efa = max_act * (1.0f - q4 - g * g);
                efa = fminf(fmaxf(efa, -50.0f), 50.0f);
                // merging vehicle IDM
                dxp = 0.1f + fmaxf(emdx, 0.0f);
                dg  = fmaxf(min_jamx + des_tgap * ego_v + ego_v * emdv / gap_denum, 0.0f);
                g   = dg / dxp;
                float ema = max_act * (1.0f - q4 - g * g);
                ema = fminf(fmaxf(ema, -50.0f), 50.0f);

                float act = fatt * efa + matt * ema;
                size_t o = (size_t)(b0 + lane) * T + t;
                out[o]          = fminf(fmaxf(act, -5.4f), 5.4f);
                out[BT + o]     = fatt;
                out[2 * BT + o] = matt;
                ego_v += act * 0.1f;
                ego_x += ego_v * 0.1f + act * 0.005f;  // new_v*0.1 + 0.5*act*0.01
            }
        }
        // no trailing barrier: next iteration's sCoefH producer == this consumer (wave 0),
        // all cross-wave LDS reuse is guarded by the phase barriers above.
    }
}

extern "C" void kernel_launch(void* const* d_in, const int* in_sizes, int n_in,
                              void* d_out, int out_size, void* d_ws, size_t ws_size,
                              hipStream_t stream) {
    const float* idm_params  = (const float*)d_in[0];
    const float* proj_belief = (const float*)d_in[1];
    const float* enc_h       = (const float*)d_in[2];
    const float* idm_s       = (const float*)d_in[3];
    const float* merger_cs   = (const float*)d_in[4];
    const float* W1 = (const float*)d_in[5];
    const float* b1 = (const float*)d_in[6];
    const float* W2 = (const float*)d_in[7];
    const float* b2 = (const float*)d_in[8];
    const float* W3 = (const float*)d_in[9];
    const float* b3 = (const float*)d_in[10];
    const float* Wf = (const float*)d_in[11];
    const float* bf = (const float*)d_in[12];
    const float* Wm = (const float*)d_in[13];
    const float* bm = (const float*)d_in[14];
    const float* sm = (const float*)d_in[15];
    const float* sv = (const float*)d_in[16];

    const int B = in_sizes[0] / 5;                 // idm_params is (B,5)
    const int T = in_sizes[3] / (B * 11);          // idm_s is (B,T,11)

    dim3 grid(B / TILE_M), block(256);
    rollout_kernel<<<grid, block, 0, stream>>>(
        idm_params, proj_belief, enc_h, idm_s, merger_cs,
        W1, b1, W2, b2, W3, b3, Wf, bf, Wm, bm, sm, sv,
        (float*)d_out, B, T);
}
